// PolySketchAttention_28613072125953
// MI455X (gfx1250) — compile-verified, matrix operand order checked by reading, not run on hardware
//
#include <hip/hip_runtime.h>

#define DEV __device__ __forceinline__

typedef __attribute__((ext_vector_type(16))) _Float16 v16h;
typedef __attribute__((ext_vector_type(8)))  _Float16 v8h;
typedef __attribute__((ext_vector_type(4)))  _Float16 v4h;
typedef __attribute__((ext_vector_type(8)))  float    v8f;
typedef __attribute__((ext_vector_type(2)))  float    v2f;

// Problem constants (from the reference)
constexpr int    Bb  = 4;
constexpr int    Ls  = 8192;
constexpr int    Dm  = 512;
constexpr int    Hh  = 8;
constexpr int    BH  = Bb * Hh;            // 32 (batch*heads)
constexpr int    Tch = Ls / 16;            // 512 chunks per sequence
constexpr int    NSEG = 16;                // segments per sequence (32 chunks each)
constexpr size_t TOFF = (size_t)BH * Ls * 64;  // floats per (B,H,L,64) tensor

// ---------------------------------------------------------------------------
// Optional CDNA5 async global->LDS copies (ASYNCcnt path)
// ---------------------------------------------------------------------------
#if defined(__has_builtin)
#if __has_builtin(__builtin_amdgcn_global_load_async_to_lds_b128) && \
    __has_builtin(__builtin_amdgcn_s_wait_asynccnt)
#define USE_ASYNC_LDS 1
#endif
#endif
#ifndef USE_ASYNC_LDS
#define USE_ASYNC_LDS 0
#endif

#if USE_ASYNC_LDS
// builtin prototype (from clang diagnostics):
//   (v4i addrspace(1)*, v4i addrspace(3)*, imm int offset, imm int cpol)
typedef int v4i __attribute__((vector_size(4 * sizeof(int))));
typedef __attribute__((address_space(1))) v4i gv4i;
typedef __attribute__((address_space(3))) v4i lv4i;
DEV void async_cp16(const float* gsrc, float* ldst) {
  // generic LDS address: low 32 bits are the LDS byte offset (flat aperture)
  __builtin_amdgcn_global_load_async_to_lds_b128(
      (gv4i*)(unsigned long long)(const void*)gsrc,
      (lv4i*)(unsigned int)(unsigned long long)(const void*)ldst,
      0, 0);
}
DEV void async_wait() { __builtin_amdgcn_s_wait_asynccnt(0); }
#endif

// ---------------------------------------------------------------------------
// WMMA helpers (CDNA5 wave32 layouts per ISA 7.12.2)
// ---------------------------------------------------------------------------
DEV v8f zero8() {
  v8f z;
#pragma unroll
  for (int i = 0; i < 8; ++i) z[i] = 0.f;
  return z;
}

DEV v8f wmmaH(v16h a, v16h b, v8f c) {
  // D = A(16x32 f16) * B(32x16 f16) + C(f32)
  return __builtin_amdgcn_wmma_f32_16x16x32_f16(false, a, false, b, (short)0, c,
                                                false, false);
}

DEV v8f wmmaF(v2f a, v2f b, v8f c) {
  // D = A(16x4 f32) * B(4x16 f32) + C(f32)
  return __builtin_amdgcn_wmma_f32_16x16x4_f32(false, a, false, b, (short)0, c,
                                               false, false);
}

// f16 A/B fragment from a row-major LDS tile [16 rows x (>=32) halfs].
// A: lane m<->row m; lanes 0-15 hold K 0..7 & 16..23, lanes 16-31 hold
// K 8..15 & 24..31.  B operand (KxN col-per-lane) uses the identical pattern
// when fed rows of B^T, which is how all our tiles are stored.
DEV v16h ldsFragH(const _Float16* tile, int stride, int lane) {
  const _Float16* p = tile + (lane & 15) * stride + ((lane & 16) ? 8 : 0);
  v8h lo = *(const v8h*)p;
  v8h hi = *(const v8h*)(p + 16);
  v16h r;
#pragma unroll
  for (int i = 0; i < 8; ++i) { r[i] = lo[i]; r[8 + i] = hi[i]; }
  return r;
}

DEV v4h cvt4(float4 v) {
  v4h h;
  h[0] = (_Float16)v.x; h[1] = (_Float16)v.y;
  h[2] = (_Float16)v.z; h[3] = (_Float16)v.w;
  return h;
}

// ---------------------------------------------------------------------------
// K1: fused QKV projection  Y = x @ W^T + b  (f16 WMMA, f32 accumulate)
//     grid (256, 24): 128-row tiles x 64-col tiles over [Q|K|V]
//     also emits deterministic per-block partial sums for the global norm
// ---------------------------------------------------------------------------
__global__ __launch_bounds__(256) void k1_qkv(
    const float* __restrict__ x,
    const float* __restrict__ wq, const float* __restrict__ bq,
    const float* __restrict__ wk, const float* __restrict__ bk,
    const float* __restrict__ wv, const float* __restrict__ bv,
    float* __restrict__ qkv, float* __restrict__ pstat)
{
  __shared__ _Float16 xb[128 * 40];
  __shared__ _Float16 wb[64 * 40];
  __shared__ float    sred[16];

  const int tid = threadIdx.x, wave = tid >> 5, lane = tid & 31;
  const int mtile = blockIdx.x;   // 0..255
  const int ntile = blockIdx.y;   // 0..23 across [Q|K|V]
  const int tsel  = ntile >> 3;   // 0=Q 1=K 2=V
  const float* W  = (tsel == 0) ? wq : (tsel == 1 ? wk : wv);
  const float* Bp = (tsel == 0) ? bq : (tsel == 1 ? bk : bv);
  const int ncol0 = (ntile & 7) * 64;

  // staging indices (loop-invariant)
  const int xrow[4] = { (tid + 0)   >> 3, (tid + 256) >> 3,
                        (tid + 512) >> 3, (tid + 768) >> 3 };
  const int xc4 = tid & 7;
  const int wrow[2] = { (tid + 0) >> 3, (tid + 256) >> 3 };

  v8f acc[4];
#pragma unroll
  for (int j = 0; j < 4; ++j) acc[j] = zero8();

  for (int k0 = 0; k0 < Dm; k0 += 32) {
    // ---- load phase: 6 independent b128 loads in flight ----
    float4 tx[4], tw[2];
#pragma unroll
    for (int q = 0; q < 4; ++q)
      tx[q] = *(const float4*)(x + (size_t)(mtile * 128 + xrow[q]) * Dm + k0 + xc4 * 4);
#pragma unroll
    for (int q = 0; q < 2; ++q)
      tw[q] = *(const float4*)(W + (size_t)(ncol0 + wrow[q]) * Dm + k0 + xc4 * 4);
    if (k0 + 32 < Dm) {
      __builtin_prefetch(x + (size_t)(mtile * 128) * Dm + k0 + 32, 0, 0);
      __builtin_prefetch(W + (size_t)ncol0 * Dm + k0 + 32, 0, 0);
    }
    // ---- convert + store phase ----
#pragma unroll
    for (int q = 0; q < 4; ++q)
      *(v4h*)(xb + xrow[q] * 40 + xc4 * 4) = cvt4(tx[q]);
#pragma unroll
    for (int q = 0; q < 2; ++q)
      *(v4h*)(wb + wrow[q] * 40 + xc4 * 4) = cvt4(tw[q]);
    __syncthreads();
    v16h a = ldsFragH(xb + wave * 16 * 40, 40, lane);
#pragma unroll
    for (int j = 0; j < 4; ++j) {
      v16h bf = ldsFragH(wb + j * 16 * 40, 40, lane);
      acc[j] = wmmaH(a, bf, acc[j]);
    }
    __syncthreads();
  }

  const int half8 = (lane & 16) ? 8 : 0;
  const int ln = lane & 15;
  float s1 = 0.f, s2 = 0.f;
#pragma unroll
  for (int j = 0; j < 4; ++j) {
    float bj = Bp[ncol0 + j * 16 + ln];
#pragma unroll
    for (int r = 0; r < 8; ++r) {
      float t = acc[j][r] + bj;
      acc[j][r] = t;
      s1 += t; s2 += t * t;
    }
  }
#pragma unroll
  for (int m = 16; m >= 1; m >>= 1) {
    s1 += __shfl_xor(s1, m, 32);
    s2 += __shfl_xor(s2, m, 32);
  }
  if (lane == 0) { sred[wave * 2] = s1; sred[wave * 2 + 1] = s2; }

  // store into (tensor, b*H+head, l, e) layout
  const int headId = ntile & 7;
  const int nrow0  = mtile * 128 + wave * 16;
  float* tb = qkv + (size_t)tsel * TOFF;
#pragma unroll
  for (int r = 0; r < 8; ++r) {
    int n = nrow0 + r + half8;
    int bidx = n >> 13, l = n & 8191;
    size_t base = ((size_t)(bidx * Hh + headId) * Ls + l) * 64;
#pragma unroll
    for (int j = 0; j < 4; ++j) tb[base + j * 16 + ln] = acc[j][r];
  }

  __syncthreads();
  if (tid == 0) {
    float a1 = 0.f, a2 = 0.f;
#pragma unroll
    for (int w = 0; w < 8; ++w) { a1 += sred[w * 2]; a2 += sred[w * 2 + 1]; }
    int lin = blockIdx.y * 256 + blockIdx.x;
    pstat[lin * 2] = a1; pstat[lin * 2 + 1] = a2;
  }
}

// ---------------------------------------------------------------------------
// K2: deterministic reduction of per-block partials -> (mu, rsqrt(var+eps))x3
// ---------------------------------------------------------------------------
__global__ __launch_bounds__(256) void k2_stats(const float* __restrict__ pstat,
                                                float* __restrict__ statsF)
{
  __shared__ float r1[256], r2[256];
  const int tid = threadIdx.x;
  for (int t = 0; t < 3; ++t) {
    float s1 = 0.f, s2 = 0.f;
#pragma unroll
    for (int q = 0; q < 8; ++q) {
      int i = t * 2048 + tid + q * 256;
      s1 += pstat[i * 2];
      s2 += pstat[i * 2 + 1];
    }
    r1[tid] = s1; r2[tid] = s2;
    __syncthreads();
    for (int off = 128; off >= 1; off >>= 1) {
      if (tid < off) { r1[tid] += r1[tid + off]; r2[tid] += r2[tid + off]; }
      __syncthreads();
    }
    if (tid == 0) {
      const float n = 16777216.0f;  // B*H*L*64
      float mu  = r1[0] / n;
      float var = r2[0] / n - mu * mu;
      statsF[t * 2]     = mu;
      statsF[t * 2 + 1] = rsqrtf(var + 1e-5f);
    }
    __syncthreads();
  }
}

// ---------------------------------------------------------------------------
// K3: global-norm + phi_p2 sketch, in place.  grid (BH*L/8, 3), one wave/row.
//     t<2: p_i = ((v-mu)*rs * sum_j((v_j-mu)*rs))^2 ; t==2: normalize only.
// ---------------------------------------------------------------------------
__global__ __launch_bounds__(256) void k3_phi(float* __restrict__ qkv,
                                              const float* __restrict__ statsF)
{
  const int wave = threadIdx.x >> 5, lane = threadIdx.x & 31;
  const int ts = blockIdx.y;
  const size_t row = (size_t)blockIdx.x * 8 + wave;
  const float mu = statsF[ts * 2];
  const float rs = statsF[ts * 2 + 1];
  float* p = qkv + (size_t)ts * TOFF + row * 64 + lane * 2;
  float2 v = *(float2*)p;
  float ax = (v.x - mu) * rs;
  float ay = (v.y - mu) * rs;
  float s = ax + ay;
#pragma unroll
  for (int m = 16; m >= 1; m >>= 1) s += __shfl_xor(s, m, 32);
  if (ts < 2) { ax *= s; ax *= ax; ay *= s; ay *= ay; }
  v.x = ax; v.y = ay;
  *(float2*)p = v;
}

// ---------------------------------------------------------------------------
// K4: per-chunk state  Hl[f][o] = sum_j pK[j][f] * Vh[j][o]   (fp32 WMMA)
//     plus ones-column Hones[f] = sum_j pK[j][f].  One wave per chunk.
// ---------------------------------------------------------------------------
__global__ __launch_bounds__(256) void k4_hl(const float* __restrict__ qkv,
                                             float* __restrict__ Hl,
                                             float* __restrict__ Hon)
{
  __shared__ float kb[8][16 * 68];
  __shared__ float vb[8][16 * 68];
  const int lane = threadIdx.x & 31, wave = threadIdx.x >> 5;
  const size_t chunk = (size_t)blockIdx.x * 8 + wave;   // 0..16383
  const int bh = (int)(chunk >> 9);
  const int t  = (int)(chunk & 511);
  const float* Kg = qkv + TOFF     + ((size_t)bh * Ls + t * 16) * 64;
  const float* Vg = qkv + 2 * TOFF + ((size_t)bh * Ls + t * 16) * 64;
  float* kl = kb[wave];
  float* vl = vb[wave];
#pragma unroll
  for (int q = 0; q < 8; ++q) {
    int idx = lane + q * 32;
    int row = idx >> 4, c4 = idx & 15;
#if USE_ASYNC_LDS
    async_cp16(Kg + row * 64 + c4 * 4, kl + row * 68 + c4 * 4);
    async_cp16(Vg + row * 64 + c4 * 4, vl + row * 68 + c4 * 4);
#else
    *(float4*)(kl + row * 68 + c4 * 4) = *(const float4*)(Kg + row * 64 + c4 * 4);
    *(float4*)(vl + row * 68 + c4 * 4) = *(const float4*)(Vg + row * 64 + c4 * 4);
#endif
  }
#if USE_ASYNC_LDS
  async_wait();
#endif
  const int ln = lane & 15;
  const int kof = (lane & 16) ? 2 : 0;
  const int half8 = (lane & 16) ? 8 : 0;
  float* outp = Hl + chunk * 4096;
#pragma unroll
  for (int ft = 0; ft < 4; ++ft) {
#pragma unroll
    for (int ot = 0; ot < 4; ++ot) {
      v8f cacc = zero8();
#pragma unroll
      for (int kk = 0; kk < 4; ++kk) {
        int j0 = kk * 4 + kof;
        v2f a; a[0] = kl[j0 * 68 + ft * 16 + ln];
               a[1] = kl[(j0 + 1) * 68 + ft * 16 + ln];
        v2f b; b[0] = vl[j0 * 68 + ot * 16 + ln];
               b[1] = vl[(j0 + 1) * 68 + ot * 16 + ln];
        cacc = wmmaF(a, b, cacc);
      }
#pragma unroll
      for (int r = 0; r < 8; ++r)
        outp[(size_t)(ft * 16 + r + half8) * 64 + ot * 16 + ln] = cacc[r];
    }
  }
  float s0 = 0.f, s1 = 0.f;
#pragma unroll
  for (int j = 0; j < 16; ++j) {
    s0 += kl[j * 68 + lane];
    s1 += kl[j * 68 + 32 + lane];
  }
  Hon[chunk * 64 + lane]      = s0;
  Hon[chunk * 64 + 32 + lane] = s1;
}

// ---------------------------------------------------------------------------
// K5a: segment sums (32 chunks per segment), 512-way parallel
// ---------------------------------------------------------------------------
__global__ __launch_bounds__(256) void k5a_segsum(const float* __restrict__ Hl,
                                                  const float* __restrict__ Hon,
                                                  float* __restrict__ Ssum,
                                                  float* __restrict__ Son)
{
  const int bh = blockIdx.x >> 4, s = blockIdx.x & 15;
  const size_t hbase = ((size_t)bh * Tch + s * 32) * 4096;
  const size_t obase = ((size_t)bh * NSEG + s) * 4096;
  for (int e = threadIdx.x; e < 4096; e += 256) {
    float a = 0.f;
    for (int c = 0; c < 32; ++c) a += Hl[hbase + (size_t)c * 4096 + e];
    Ssum[obase + e] = a;
  }
  if (threadIdx.x < 64) {
    const size_t h1 = ((size_t)bh * Tch + s * 32) * 64;
    float a = 0.f;
    for (int c = 0; c < 32; ++c) a += Hon[h1 + c * 64 + threadIdx.x];
    Son[((size_t)bh * NSEG + s) * 64 + threadIdx.x] = a;
  }
}

// ---------------------------------------------------------------------------
// K5b: exclusive scan over the 16 segment sums per (b,head)
// ---------------------------------------------------------------------------
__global__ __launch_bounds__(256) void k5b_scan(const float* __restrict__ Ssum,
                                                const float* __restrict__ Son,
                                                float* __restrict__ Zseg,
                                                float* __restrict__ Zon)
{
  const int bh = blockIdx.x;
  for (int e = threadIdx.x; e < 4096; e += 256) {
    float run = 0.f;
    for (int s = 0; s < NSEG; ++s) {
      size_t i = ((size_t)bh * NSEG + s) * 4096 + e;
      Zseg[i] = run;
      run += Ssum[i];
    }
  }
  if (threadIdx.x < 64) {
    float run = 0.f;
    for (int s = 0; s < NSEG; ++s) {
      size_t i = ((size_t)bh * NSEG + s) * 64 + threadIdx.x;
      Zon[i] = run;
      run += Son[i];
    }
  }
}

// ---------------------------------------------------------------------------
// K6: main causal output. One block (4 waves) per (bh, segment); Z in LDS.
//     out = (tril(pQ pK^T) @ V  +  pQ @ Z) / (1 + rowsum(tril S) + pQ.Zones)
// ---------------------------------------------------------------------------
__global__ __launch_bounds__(128) void k6_attn(const float* __restrict__ qkv,
                                               const float* __restrict__ Hl,
                                               const float* __restrict__ Hon,
                                               const float* __restrict__ Zseg,
                                               const float* __restrict__ Zon,
                                               float* __restrict__ attn)
{
  __shared__ float Zl[64 * 68];
  __shared__ float zo[64];
  __shared__ float qb[16 * 68], kb[16 * 68], vb[16 * 68];
  __shared__ float Sl[16 * 20];
  __shared__ float rsl[16], dtl[16];

  const int tid = threadIdx.x, wave = tid >> 5, lane = tid & 31;
  const int bh = blockIdx.x >> 4, seg = blockIdx.x & 15;
  const int half8 = (lane & 16) ? 8 : 0;
  const int ln = lane & 15;
  const int kof = (lane & 16) ? 2 : 0;

  for (int e = tid; e < 4096; e += 128)
    Zl[(e >> 6) * 68 + (e & 63)] = Zseg[((size_t)(bh * NSEG + seg)) * 4096 + e];
  if (tid < 64) zo[tid] = Zon[(bh * NSEG + seg) * 64 + tid];

  const float* Qg = qkv;
  const float* Kg = qkv + TOFF;
  const float* Vg = qkv + 2 * TOFF;

  for (int c = 0; c < 32; ++c) {
    const int t = seg * 32 + c;
    __syncthreads();                       // Z / previous iteration complete
#pragma unroll
    for (int q = 0; q < 2; ++q) {          // stage pQ, pK, Vh chunk (16x64 each)
      int idx = tid + q * 128;
      int row = idx >> 4, c4 = idx & 15;
      size_t g = ((size_t)bh * Ls + t * 16 + row) * 64 + c4 * 4;
#if USE_ASYNC_LDS
      async_cp16(Qg + g, qb + row * 68 + c4 * 4);
      async_cp16(Kg + g, kb + row * 68 + c4 * 4);
      async_cp16(Vg + g, vb + row * 68 + c4 * 4);
#else
      *(float4*)(qb + row * 68 + c4 * 4) = *(const float4*)(Qg + g);
      *(float4*)(kb + row * 68 + c4 * 4) = *(const float4*)(Kg + g);
      *(float4*)(vb + row * 68 + c4 * 4) = *(const float4*)(Vg + g);
#endif
    }
#if USE_ASYNC_LDS
    async_wait();                          // LDS writes landed before barrier
#endif
    __syncthreads();

    // inter-chunk: oacc = pQ @ Z for this wave's 16-wide o-tile
    const int ot = wave;
    v8f oacc = zero8();
#pragma unroll
    for (int kk = 0; kk < 16; ++kk) {
      int f0 = kk * 4 + kof;
      v2f a; a[0] = qb[ln * 68 + f0];       a[1] = qb[ln * 68 + f0 + 1];
      v2f b; b[0] = Zl[f0 * 68 + ot * 16 + ln];
             b[1] = Zl[(f0 + 1) * 68 + ot * 16 + ln];
      oacc = wmmaF(a, b, oacc);
    }

    if (wave == 0) {                       // wave-uniform branch, EXEC all ones
      v8f s = zero8();                     // S = pQ pK^T (16x16)
#pragma unroll
      for (int kk = 0; kk < 16; ++kk) {
        int f0 = kk * 4 + kof;
        v2f a; a[0] = qb[ln * 68 + f0]; a[1] = qb[ln * 68 + f0 + 1];
        v2f b; b[0] = kb[ln * 68 + f0]; b[1] = kb[ln * 68 + f0 + 1];
        s = wmmaF(a, b, s);
      }
#pragma unroll
      for (int r = 0; r < 8; ++r) {        // tril mask + rowsums + share S
        int i = r + half8;
        if (ln > i) s[r] = 0.f;
        float v = s[r];
#pragma unroll
        for (int m = 8; m >= 1; m >>= 1) v += __shfl_xor(v, m, 16);
        if (ln == 0) rsl[i] = v;
        Sl[i * 20 + ln] = s[r];
      }
      if (lane < 16) {                     // Dt = 1 + rowsum + pQ . Zones
        float z = 0.f;
#pragma unroll
        for (int f = 0; f < 64; ++f) z += qb[lane * 68 + f] * zo[f];
        dtl[lane] = 1.0f + rsl[lane] + z;
      }
    }
    __syncthreads();

    // intra-chunk: oacc += S @ V
#pragma unroll
    for (int kk = 0; kk < 4; ++kk) {
      int j0 = kk * 4 + kof;
      v2f a; a[0] = Sl[ln * 20 + j0]; a[1] = Sl[ln * 20 + j0 + 1];
      v2f b; b[0] = vb[j0 * 68 + ot * 16 + ln];
             b[1] = vb[(j0 + 1) * 68 + ot * 16 + ln];
      oacc = wmmaF(a, b, oacc);
    }

    // divide by denominator and store
    size_t ob = ((size_t)bh * Ls + t * 16) * 64;
#pragma unroll
    for (int r = 0; r < 8; ++r) {
      int i = r + half8;
      attn[ob + (size_t)i * 64 + ot * 16 + ln] = oacc[r] / dtl[i];
    }

    // advance running state: Z += Hl[chunk], Zones += Hones[chunk]
    size_t hb = ((size_t)bh * Tch + t) * 4096;
    for (int e = tid; e < 4096; e += 128)
      Zl[(e >> 6) * 68 + (e & 63)] += Hl[hb + e];
    if (tid < 64) zo[tid] += Hon[((size_t)bh * Tch + t) * 64 + tid];
  }
}

// ---------------------------------------------------------------------------
// K7: output projection  out = o @ wo^T + bo  (f16 WMMA, f32 accumulate)
//     A rows gathered from the (B,H,L,64) attention buffer (implicit
//     transpose(0,2,1,3).reshape).  grid (256, 8).
// ---------------------------------------------------------------------------
__global__ __launch_bounds__(256) void k7_out(const float* __restrict__ attn,
                                              const float* __restrict__ wo,
                                              const float* __restrict__ bo,
                                              float* __restrict__ out)
{
  __shared__ _Float16 ab[128 * 40];
  __shared__ _Float16 wb[64 * 40];
  const int tid = threadIdx.x, wave = tid >> 5, lane = tid & 31;
  const int mtile = blockIdx.x;
  const int ntile = blockIdx.y;
  const int ncol0 = ntile * 64;

  const int xrow[4] = { (tid + 0)   >> 3, (tid + 256) >> 3,
                        (tid + 512) >> 3, (tid + 768) >> 3 };
  const int xc4 = tid & 7;
  const int wrow[2] = { (tid + 0) >> 3, (tid + 256) >> 3 };

  v8f acc[4];
#pragma unroll
  for (int j = 0; j < 4; ++j) acc[j] = zero8();

  for (int k0 = 0; k0 < Dm; k0 += 32) {
    // ---- load phase ----
    float4 tx[4], tw[2];
#pragma unroll
    for (int q = 0; q < 4; ++q) {
      int n = mtile * 128 + xrow[q];
      int bidx = n >> 13, l = n & 8191;
      int hh = k0 >> 6, e0 = (k0 & 63) + xc4 * 4;
      tx[q] = *(const float4*)(attn + ((size_t)(bidx * Hh + hh) * Ls + l) * 64 + e0);
    }
#pragma unroll
    for (int q = 0; q < 2; ++q)
      tw[q] = *(const float4*)(wo + (size_t)(ncol0 + wrow[q]) * Dm + k0 + xc4 * 4);
    // ---- convert + store phase ----
#pragma unroll
    for (int q = 0; q < 4; ++q)
      *(v4h*)(ab + xrow[q] * 40 + xc4 * 4) = cvt4(tx[q]);
#pragma unroll
    for (int q = 0; q < 2; ++q)
      *(v4h*)(wb + wrow[q] * 40 + xc4 * 4) = cvt4(tw[q]);
    __syncthreads();
    v16h a = ldsFragH(ab + wave * 16 * 40, 40, lane);
#pragma unroll
    for (int j = 0; j < 4; ++j) {
      v16h bf = ldsFragH(wb + j * 16 * 40, 40, lane);
      acc[j] = wmmaH(a, bf, acc[j]);
    }
    __syncthreads();
  }

  const int half8 = (lane & 16) ? 8 : 0;
  const int ln = lane & 15;
  const int nrow0 = mtile * 128 + wave * 16;
#pragma unroll
  for (int j = 0; j < 4; ++j) {
    float bj = bo[ncol0 + j * 16 + ln];
#pragma unroll
    for (int r = 0; r < 8; ++r)
      out[(size_t)(nrow0 + r + half8) * Dm + ncol0 + j * 16 + ln] = acc[j][r] + bj;
  }
}

// ---------------------------------------------------------------------------
// Host launcher
// ---------------------------------------------------------------------------
extern "C" void kernel_launch(void* const* d_in, const int* in_sizes, int n_in,
                              void* d_out, int out_size, void* d_ws, size_t ws_size,
                              hipStream_t stream)
{
  (void)in_sizes; (void)n_in; (void)out_size; (void)ws_size;
  const float* x  = (const float*)d_in[0];
  const float* wq = (const float*)d_in[1];
  const float* bq = (const float*)d_in[2];
  const float* wk = (const float*)d_in[3];
  const float* bk = (const float*)d_in[4];
  const float* wv = (const float*)d_in[5];
  const float* bv = (const float*)d_in[6];
  const float* wo = (const float*)d_in[7];
  const float* bo = (const float*)d_in[8];
  float* out = (float*)d_out;
  float* ws  = (float*)d_ws;

  float* qkv    = ws;                                   // 3*TOFF floats
  float* attn   = qkv  + 3 * TOFF;                      // TOFF
  float* Hl     = attn + TOFF;                          // BH*Tch*4096
  float* Hon    = Hl   + (size_t)BH * Tch * 4096;       // BH*Tch*64
  float* Ssum   = Hon  + (size_t)BH * Tch * 64;         // BH*NSEG*4096
  float* Son    = Ssum + (size_t)BH * NSEG * 4096;      // BH*NSEG*64
  float* Zseg   = Son  + (size_t)BH * NSEG * 64;        // BH*NSEG*4096
  float* Zon    = Zseg + (size_t)BH * NSEG * 4096;      // BH*NSEG*64
  float* pstat  = Zon  + (size_t)BH * NSEG * 64;        // 12288
  float* statsF = pstat + 12288;                        // 8

  k1_qkv   <<<dim3(256, 24),       256, 0, stream>>>(x, wq, bq, wk, bk, wv, bv, qkv, pstat);
  k2_stats <<<1,                   256, 0, stream>>>(pstat, statsF);
  k3_phi   <<<dim3(BH * Ls / 8, 3),256, 0, stream>>>(qkv, statsF);
  k4_hl    <<<(BH * Tch) / 8,      256, 0, stream>>>(qkv, Hl, Hon);
  k5a_segsum<<<BH * NSEG,          256, 0, stream>>>(Hl, Hon, Ssum, Son);
  k5b_scan <<<BH,                  256, 0, stream>>>(Ssum, Son, Zseg, Zon);
  k6_attn  <<<BH * NSEG,           128, 0, stream>>>(qkv, Hl, Hon, Zseg, Zon, attn);
  k7_out   <<<dim3(256, 8),        256, 0, stream>>>(attn, wo, bo, out);
}
